// MLAttention_45432164057336
// MI455X (gfx1250) — matrix operand order repaired and checked by reading, not compiled, not run on hardware
//
#include <hip/hip_runtime.h>
#include <hip/hip_bf16.h>
#include <stdint.h>

typedef __attribute__((ext_vector_type(16))) _Float16 v16h;
typedef __attribute__((ext_vector_type(8)))  float    v8f;

#define B_SZ 2
#define T_SZ 2048
#define H_SZ 16
#define DM   2048
#define DQ   682      // D_MODEL // 3
#define DQP  704      // DQ padded to multiple of 32
#define DKV  1024
#define DKVR 1088     // DKV + D_QK_R
#define DUKV 3072     // D_MODEL + N_HEAD*D_QK_N
#define DH   128

// ---------------- WMMA fragment helpers (ISA 7.12.2 layouts, wave32) -------

struct AF { union { v16h v; unsigned u[8]; }; };

__device__ __forceinline__ v8f vzero8() {
  v8f z = {0.f,0.f,0.f,0.f,0.f,0.f,0.f,0.f};
  return z;
}

// A-matrix 16x32 f16: lane = row M (mod 16), half selects K groups.
// VGPR j<4: K = 2j + 8*half ; j>=4: K = 2(j-4) + 16 + 8*half.
// Dword indices are contiguous -> compiler emits 2x ds_load_b128.
__device__ __forceinline__ void load_a_frag(AF& f, const _Float16* s, int rowbase,
                                            int stride, int kofs, int l16, int hf) {
  const unsigned* p = (const unsigned*)(s + (size_t)(rowbase + l16) * stride + kofs);
  #pragma unroll
  for (int j = 0; j < 8; ++j) {
    int kb = (j < 4) ? (2*j + 8*hf) : (2*(j-4) + 16 + 8*hf);
    f.u[j] = p[kb >> 1];
  }
}

// B-matrix 32x16 f16 where LDS holds rows of N (i.e. B[k][n] = lds[n][k]):
// lane = col N, VGPR j<4: K = 2j + 8*half ; j>=4: K = 2(j-4) + 16 + 8*half.
__device__ __forceinline__ void load_b_frag_nk(AF& f, const _Float16* s, int rowbase,
                                               int stride, int kofs, int l16, int hf) {
  const unsigned* p = (const unsigned*)(s + (size_t)(rowbase + l16) * stride + kofs);
  #pragma unroll
  for (int j = 0; j < 8; ++j) {
    int kb = (j < 4) ? (2*j + 8*hf) : (2*(j-4) + 16 + 8*hf);
    f.u[j] = p[kb >> 1];
  }
}

__device__ __forceinline__ v8f wmma_f16(const AF& a, const AF& b, v8f c) {
  return __builtin_amdgcn_wmma_f32_16x16x32_f16(false, a.v, false, b.v,
                                                (short)0, c, false, false);
}

// ---------------- weight pre-conversion: f32 (NxK) -> f16 (Npad x Kpad) ----

__global__ void pad_weight_f16_kernel(const float* __restrict__ W, int N, int K,
                                      _Float16* __restrict__ Wp, int Kp) {
  const int row = blockIdx.x;                 // grid.x == Npad
  _Float16* dst = Wp + (size_t)row * Kp;
  if (row < N) {
    const float* src = W + (size_t)row * K;
    for (int k = threadIdx.x; k < Kp; k += blockDim.x)
      dst[k] = (k < K) ? (_Float16)src[k] : (_Float16)0.f;
  } else {
    for (int k = threadIdx.x; k < Kp; k += blockDim.x) dst[k] = (_Float16)0.f;
  }
}

// ---- GEMM: C(f32) = A(f16, M x Kpad) * Wp(f16, Npad x Kpad)^T + bias ------
// 256 threads (8 waves, 4x2), 128x128 tile, K-slabs of 32, double-buffered LDS.

#define GT_M 128
#define GT_N 128
#define GT_K 32
#define GS   56   // LDS row stride in halfs: 112B, 16B-aligned, conflict-free

__global__ __launch_bounds__(256)
void gemm_ff_wmma(const _Float16* __restrict__ A,
                  const _Float16* __restrict__ Wp,
                  const float* __restrict__ bias,
                  float* __restrict__ C, int ldc,
                  int N, int Kpad) {
  __shared__ alignas(16) _Float16 sA[2][GT_M * GS];
  __shared__ alignas(16) _Float16 sB[2][GT_N * GS];
  const int tid = threadIdx.x;
  const int lane = tid & 31, wid = tid >> 5;
  const int wm = wid & 3, wn = wid >> 2;          // 4 x 2 wave grid
  const int hf = lane >> 4, l16 = lane & 15;
  const int mb = blockIdx.y * GT_M;
  const int nb = blockIdx.x * GT_N;

  v8f acc[2][4];
  #pragma unroll
  for (int i = 0; i < 2; ++i)
    #pragma unroll
    for (int j = 0; j < 4; ++j) acc[i][j] = vzero8();

  const int srow = tid >> 1;          // 0..127
  const int scol = (tid & 1) << 4;    // 0 or 16
  const _Float16* gA = A  + (size_t)(mb + srow) * Kpad + scol;
  const _Float16* gB = Wp + (size_t)(nb + srow) * Kpad + scol;
  const int soff = srow * GS + scol;

  { // prologue: stage slab 0 into buffer 0
    uint4 a0 = ((const uint4*)gA)[0], a1 = ((const uint4*)gA)[1];
    uint4 b0 = ((const uint4*)gB)[0], b1 = ((const uint4*)gB)[1];
    uint4* pa = (uint4*)(sA[0] + soff);
    pa[0] = a0; pa[1] = a1;
    uint4* pb = (uint4*)(sB[0] + soff);
    pb[0] = b0; pb[1] = b1;
  }

  const int nk = Kpad / GT_K;
  for (int kt = 0; kt < nk; ++kt) {
    const int cur = kt & 1;
    __syncthreads();
    // prefetch next slab into registers (overlaps with WMMA below)
    uint4 na0, na1, nb0, nb1;
    const bool more = (kt + 1) < nk;
    if (more) {
      const _Float16* pga = gA + (size_t)(kt + 1) * GT_K;
      const _Float16* pgb = gB + (size_t)(kt + 1) * GT_K;
      na0 = ((const uint4*)pga)[0]; na1 = ((const uint4*)pga)[1];
      nb0 = ((const uint4*)pgb)[0]; nb1 = ((const uint4*)pgb)[1];
    }
    AF af[2]; AF bf[4];
    #pragma unroll
    for (int i = 0; i < 2; ++i) load_a_frag(af[i], sA[cur], wm*32 + i*16, GS, 0, l16, hf);
    #pragma unroll
    for (int j = 0; j < 4; ++j) load_b_frag_nk(bf[j], sB[cur], wn*64 + j*16, GS, 0, l16, hf);
    #pragma unroll
    for (int i = 0; i < 2; ++i)
      #pragma unroll
      for (int j = 0; j < 4; ++j)
        acc[i][j] = wmma_f16(af[i], bf[j], acc[i][j]);
    if (more) {
      uint4* pa = (uint4*)(sA[cur ^ 1] + soff);
      pa[0] = na0; pa[1] = na1;
      uint4* pb = (uint4*)(sB[cur ^ 1] + soff);
      pb[0] = nb0; pb[1] = nb1;
    }
  }

  // epilogue: C-layout VGPR r -> row (r + 8*half), col = l16
  #pragma unroll
  for (int i = 0; i < 2; ++i) {
    const int rb = mb + wm*32 + i*16;
    #pragma unroll
    for (int j = 0; j < 4; ++j) {
      const int cb = nb + wn*64 + j*16 + l16;
      if (cb < N) {
        const float bv = bias[cb];
        #pragma unroll
        for (int r = 0; r < 8; ++r)
          C[(size_t)(rb + r + 8*hf) * ldc + cb] = acc[i][j][r] + bv;
      }
    }
  }
}

// ---------------- LayerNorm (one wave per row) + optional f16 tail copy ----

__global__ __launch_bounds__(128)
void layernorm_kernel(const float* __restrict__ in, int in_stride, int width,
                      const float* __restrict__ g, const float* __restrict__ bb,
                      _Float16* __restrict__ out, int out_stride,
                      int tail_w, _Float16* __restrict__ tail_out, int rows) {
  const int lane = threadIdx.x & 31;
  const int row = blockIdx.x * (blockDim.x >> 5) + (threadIdx.x >> 5);
  if (row >= rows) return;
  const float* rp = in + (size_t)row * in_stride;
  float s = 0.f;
  for (int i = lane; i < width; i += 32) s += rp[i];
  #pragma unroll
  for (int m = 16; m >= 1; m >>= 1) s += __shfl_xor(s, m, 32);
  const float mu = s / (float)width;
  float v = 0.f;
  for (int i = lane; i < width; i += 32) { const float d = rp[i] - mu; v += d * d; }
  #pragma unroll
  for (int m = 16; m >= 1; m >>= 1) v += __shfl_xor(v, m, 32);
  const float rstd = rsqrtf(v / (float)width + 1e-5f);
  for (int i = lane; i < out_stride; i += 32) {
    const float o = (i < width) ? (rp[i] - mu) * rstd * g[i] + bb[i] : 0.f;
    out[(size_t)row * out_stride + i] = (_Float16)o;
  }
  for (int i = lane; i < tail_w; i += 32)
    tail_out[(size_t)row * tail_w + i] = (_Float16)rp[width + i];
}

// ---------------- misc: f32 -> f16 bulk convert ----------------------------

__global__ void f32_to_f16_kernel(const float* __restrict__ in,
                                  _Float16* __restrict__ out, int n) {
  int i = blockIdx.x * blockDim.x + threadIdx.x;
  const int stride = gridDim.x * blockDim.x;
  for (; i < n; i += stride) out[i] = (_Float16)in[i];
}

// ---------------- pack q: rotary (position = head index!) -> (B,H,T,128) ---

__global__ __launch_bounds__(128)
void pack_q_kernel(const float* __restrict__ q_raw, _Float16* __restrict__ Qf) {
  const int t = blockIdx.x, bh = blockIdx.y;
  const int h = bh & (H_SZ - 1);
  const int b = bh >> 4;
  const int d = threadIdx.x;
  const size_t src = ((size_t)(b * T_SZ + t)) * DM + (size_t)h * DH;
  float val;
  if (d < 64) {
    val = q_raw[src + d];
  } else {
    const int i = (d - 64) & 31;
    const float x1 = q_raw[src + 64 + i];
    const float x2 = q_raw[src + 96 + i];
    const float inv = __powf(10000.f, -(float)i * (1.f / 32.f));
    const float ang = (float)h * inv;   // reference uses shape[1]==H as position
    float c, s;
    __sincosf(ang, &c, &s);
    val = (d < 96) ? (x1 * c + x2 * s) : (x2 * c - x1 * s);
  }
  Qf[((size_t)bh * T_SZ + t) * DH + d] = (_Float16)val;
}

// ---------------- pack kv: k_full (B,H,T,128) and V^T (B,H,128,T) ----------

__global__ __launch_bounds__(128)
void pack_kv_kernel(const float* __restrict__ kv_raw, const _Float16* __restrict__ kr,
                    _Float16* __restrict__ Kf, _Float16* __restrict__ Vt) {
  const int t = blockIdx.x, bh = blockIdx.y;
  const int h = bh & (H_SZ - 1);
  const int b = bh >> 4;
  const int d = threadIdx.x;
  const size_t row = (size_t)(b * T_SZ + t);
  // k_r rotary is identity (position axis has extent 1 in the reference)
  const float kval = (d < 64) ? kv_raw[row * DUKV + h * 192 + d]
                              : (float)kr[row * 64 + (d - 64)];
  Kf[((size_t)bh * T_SZ + t) * DH + d] = (_Float16)kval;
  const float vval = kv_raw[row * DUKV + h * 192 + 64 + d];
  Vt[((size_t)bh * DH + d) * T_SZ + t] = (_Float16)vval;
}

// ---------------- flash attention: 4 waves/block, 16 queries per wave ------

#define SK 136   // [key|qrow][d] stride (halfs): 272B rows, 16B aligned
#define SV 56    // V^T tile [d][key] stride
#define SP 36    // wave-private P slab stride (even -> u32 frag loads ok)

__global__ __launch_bounds__(128)
void flash_attn_kernel(const _Float16* __restrict__ Qf, const _Float16* __restrict__ Kf,
                       const _Float16* __restrict__ Vt, _Float16* __restrict__ Y) {
  __shared__ alignas(16) _Float16 sQ[64 * SK];
  __shared__ alignas(16) _Float16 sK[32 * SK];
  __shared__ alignas(16) _Float16 sV[DH * SV];
  __shared__ alignas(16) _Float16 sP[4 * 16 * SP];

  const int tid = threadIdx.x;
  const int lane = tid & 31, w = tid >> 5;
  const int hf = lane >> 4, l16 = lane & 15;
  const int tq = blockIdx.x;            // 64-query tile
  const int bh = blockIdx.y;
  const size_t qoff = ((size_t)bh * T_SZ + (size_t)tq * 64) * DH;
  const size_t koff = (size_t)bh * T_SZ * DH;
  const size_t voff = (size_t)bh * DH * T_SZ;

  // stage the whole 64x128 Q tile
  for (int idx = tid; idx < 64 * 4; idx += 128) {
    const int r = idx >> 2, c = (idx & 3) << 5;
    const uint4* gp = (const uint4*)(Qf + qoff + (size_t)r * DH + c);
    uint4 d0 = gp[0], d1 = gp[1], d2 = gp[2], d3 = gp[3];
    uint4* sp = (uint4*)(sQ + r * SK + c);
    sp[0] = d0; sp[1] = d1; sp[2] = d2; sp[3] = d3;
  }
  __syncthreads();
  AF qf[4];
  #pragma unroll
  for (int dk = 0; dk < 4; ++dk) load_a_frag(qf[dk], sQ, w * 16, SK, dk * 32, l16, hf);

  v8f o[8];
  #pragma unroll
  for (int dt = 0; dt < 8; ++dt) o[dt] = vzero8();
  float mrow[8], lrow[8];
  #pragma unroll
  for (int r = 0; r < 8; ++r) { mrow[r] = -1e30f; lrow[r] = 0.f; }

  const int qg_base = tq * 64 + w * 16;
  const int nkt = (tq + 1) * 2;               // causal: keys up to block's max query
  const float scale = 0.08838834764831845f;   // 1/sqrt(128)
  _Float16* sPw = sP + w * 16 * SP;

  for (int kt = 0; kt < nkt; ++kt) {
    const int k0 = kt * 32;
    __syncthreads();
    // stage K tile: 32 keys x 128d
    for (int idx = tid; idx < 32 * 4; idx += 128) {
      const int r = idx >> 2, c = (idx & 3) << 5;
      const uint4* gp = (const uint4*)(Kf + koff + (size_t)(k0 + r) * DH + c);
      uint4 d0 = gp[0], d1 = gp[1], d2 = gp[2], d3 = gp[3];
      uint4* sp = (uint4*)(sK + r * SK + c);
      sp[0] = d0; sp[1] = d1; sp[2] = d2; sp[3] = d3;
    }
    // stage V^T tile: 128d x 32 keys (one row per thread)
    {
      const uint4* gp = (const uint4*)(Vt + voff + (size_t)tid * T_SZ + k0);
      uint4 d0 = gp[0], d1 = gp[1], d2 = gp[2], d3 = gp[3];
      uint4* sp = (uint4*)(sV + tid * SV);
      sp[0] = d0; sp[1] = d1; sp[2] = d2; sp[3] = d3;
    }
    __syncthreads();
    if (k0 <= qg_base + 15) {
      // S = Q K^T : two 16x16 score tiles (keys k0..+15, k0+16..+31), K=128
      v8f st0 = vzero8(), st1 = vzero8();
      #pragma unroll
      for (int dk = 0; dk < 4; ++dk) {
        AF kb0, kb1;
        load_b_frag_nk(kb0, sK, 0,  SK, dk * 32, l16, hf);
        load_b_frag_nk(kb1, sK, 16, SK, dk * 32, l16, hf);
        st0 = wmma_f16(qf[dk], kb0, st0);
        st1 = wmma_f16(qf[dk], kb1, st1);
      }
      // causal mask + scale; row max over the 16-lane half group
      float rowmax[8];
      #pragma unroll
      for (int r = 0; r < 8; ++r) {
        const int qg = qg_base + r + 8 * hf;
        const float s0 = (k0 + l16      <= qg) ? st0[r] * scale : -1e30f;
        const float s1 = (k0 + 16 + l16 <= qg) ? st1[r] * scale : -1e30f;
        st0[r] = s0; st1[r] = s1;
        rowmax[r] = fmaxf(s0, s1);
      }
      #pragma unroll
      for (int m = 8; m >= 1; m >>= 1)
        #pragma unroll
        for (int r = 0; r < 8; ++r)
          rowmax[r] = fmaxf(rowmax[r], __shfl_xor(rowmax[r], m, 32));
      float alpha[8], rsum[8];
      #pragma unroll
      for (int r = 0; r < 8; ++r) {
        const float mn = fmaxf(mrow[r], rowmax[r]);
        alpha[r] = __expf(mrow[r] - mn);
        mrow[r] = mn;
        const float p0 = __expf(st0[r] - mn);
        const float p1 = __expf(st1[r] - mn);
        st0[r] = p0; st1[r] = p1;
        rsum[r] = p0 + p1;
      }
      #pragma unroll
      for (int m = 8; m >= 1; m >>= 1)
        #pragma unroll
        for (int r = 0; r < 8; ++r)
          rsum[r] += __shfl_xor(rsum[r], m, 32);
      #pragma unroll
      for (int r = 0; r < 8; ++r) lrow[r] = lrow[r] * alpha[r] + rsum[r];
      #pragma unroll
      for (int dt = 0; dt < 8; ++dt)
        #pragma unroll
        for (int r = 0; r < 8; ++r) o[dt][r] *= alpha[r];
      // C-layout -> A-layout via wave-private LDS slab (same-wave LDS is in-order)
      #pragma unroll
      for (int r = 0; r < 8; ++r) {
        sPw[(r + 8 * hf) * SP + l16]      = (_Float16)st0[r];
        sPw[(r + 8 * hf) * SP + 16 + l16] = (_Float16)st1[r];
      }
      AF pf;
      load_a_frag(pf, sPw, 0, SP, 0, l16, hf);
      #pragma unroll
      for (int dt = 0; dt < 8; ++dt) {
        AF vb;
        load_b_frag_nk(vb, sV, dt * 16, SV, 0, l16, hf);
        o[dt] = wmma_f16(pf, vb, o[dt]);
      }
    }
  }
  // normalize and write y in (B, T, H*128) f16 layout for the final GEMM
  const int b_ = bh >> 4, h_ = bh & (H_SZ - 1);
  float invl[8];
  #pragma unroll
  for (int r = 0; r < 8; ++r) invl[r] = 1.f / lrow[r];
  #pragma unroll
  for (int dt = 0; dt < 8; ++dt)
    #pragma unroll
    for (int r = 0; r < 8; ++r) {
      const int qg = qg_base + r + 8 * hf;
      Y[((size_t)(b_ * T_SZ + qg)) * DM + (size_t)h_ * DH + dt * 16 + l16] =
          (_Float16)(o[dt][r] * invl[r]);
    }
}

// ---------------- host launcher --------------------------------------------

extern "C" void kernel_launch(void* const* d_in, const int* in_sizes, int n_in,
                              void* d_out, int out_size, void* d_ws, size_t ws_size,
                              hipStream_t stream) {
  const float* x      = (const float*)d_in[0];
  const float* Wdq_w  = (const float*)d_in[1];
  const float* Wdq_b  = (const float*)d_in[2];
  const float* qn_g   = (const float*)d_in[3];
  const float* qn_b   = (const float*)d_in[4];
  const float* Wuq_w  = (const float*)d_in[5];
  const float* Wuq_b  = (const float*)d_in[6];
  const float* Wdkv_w = (const float*)d_in[7];
  const float* Wdkv_b = (const float*)d_in[8];
  const float* kvn_g  = (const float*)d_in[9];
  const float* kvn_b  = (const float*)d_in[10];
  const float* Wukv_w = (const float*)d_in[11];
  const float* Wukv_b = (const float*)d_in[12];
  const float* Wo_w   = (const float*)d_in[13];
  const float* Wo_b   = (const float*)d_in[14];
  float* out = (float*)d_out;
  (void)in_sizes; (void)n_in; (void)out_size; (void)ws_size;

  char* ws = (char*)d_ws;
  size_t off = 0;
  auto carve = [&](size_t bytes) -> char* {
    char* p = ws + off;
    off += (bytes + 255) & ~(size_t)255;
    return p;
  };
  const size_t R = (size_t)B_SZ * T_SZ;  // 4096 rows

  // activations / intermediates
  _Float16* X16  = (_Float16*)carve(R * DM  * 2);  // x in f16
  _Float16* CQ16 = (_Float16*)carve(R * DQP * 2);  // LN(c_q), zero-padded K
  _Float16* QF   = (_Float16*)carve(R * DM  * 2);  // q_full (B,H,T,128)
  _Float16* KF   = (_Float16*)carve(R * DM  * 2);  // k_full (B,H,T,128)
  _Float16* VT   = (_Float16*)carve(R * DM  * 2);  // v^T    (B,H,128,T)
  _Float16* KVN  = (_Float16*)carve(R * DKV * 2);  // LN(kv_n)
  _Float16* KR   = (_Float16*)carve(R * 64  * 2);  // k_r tail
  _Float16* Y16  = (_Float16*)carve(R * DM  * 2);  // attention output
  float*    S    = (float*)carve(R * DUKV * 4);    // shared f32 GEMM scratch

  // padded f16 weights: Npad x Kpad, zero-filled
  const int NP_DQ = 768, NP_DKV = 1152;            // 682->768, 1088->1152
  _Float16* Wdq_p  = (_Float16*)carve((size_t)NP_DQ  * DM  * 2);
  _Float16* Wuq_p  = (_Float16*)carve((size_t)DM     * DQP * 2);
  _Float16* Wdkv_p = (_Float16*)carve((size_t)NP_DKV * DM  * 2);
  _Float16* Wukv_p = (_Float16*)carve((size_t)DUKV   * DKV * 2);
  _Float16* Wo_p   = (_Float16*)carve((size_t)DM     * DM  * 2);

  const dim3 blk256(256), blk128(128);

  // one-time (per launch) conversions
  f32_to_f16_kernel<<<2048, blk256, 0, stream>>>(x, X16, (int)(R * DM));
  pad_weight_f16_kernel<<<NP_DQ,  blk256, 0, stream>>>(Wdq_w,  DQ,   DM,  Wdq_p,  DM);
  pad_weight_f16_kernel<<<DM,     blk256, 0, stream>>>(Wuq_w,  DM,   DQ,  Wuq_p,  DQP);
  pad_weight_f16_kernel<<<NP_DKV, blk256, 0, stream>>>(Wdkv_w, DKVR, DM,  Wdkv_p, DM);
  pad_weight_f16_kernel<<<DUKV,   blk256, 0, stream>>>(Wukv_w, DUKV, DKV, Wukv_p, DKV);
  pad_weight_f16_kernel<<<DM,     blk256, 0, stream>>>(Wo_w,   DM,   DM,  Wo_p,   DM);

  // c_q = x @ Wdq^T + b
  gemm_ff_wmma<<<dim3(NP_DQ / 128, (unsigned)(R / 128)), blk256, 0, stream>>>(
      X16, Wdq_p, Wdq_b, S, DQ, DQ, DM);
  layernorm_kernel<<<(int)(R / 4), blk128, 0, stream>>>(
      S, DQ, DQ, qn_g, qn_b, CQ16, DQP, 0, (_Float16*)nullptr, (int)R);

  // q = LN(c_q) @ Wuq^T + b  (K padded 682 -> 704, zeros contribute nothing)
  gemm_ff_wmma<<<dim3(DM / 128, (unsigned)(R / 128)), blk256, 0, stream>>>(
      CQ16, Wuq_p, Wuq_b, S, DM, DM, DQP);
  pack_q_kernel<<<dim3(T_SZ, B_SZ * H_SZ), blk128, 0, stream>>>(S, QF);

  // c_kv = x @ Wdkv^T + b ; split into LN(kv_n) and k_r tail
  gemm_ff_wmma<<<dim3(NP_DKV / 128, (unsigned)(R / 128)), blk256, 0, stream>>>(
      X16, Wdkv_p, Wdkv_b, S, DKVR, DKVR, DM);
  layernorm_kernel<<<(int)(R / 4), blk128, 0, stream>>>(
      S, DKVR, DKV, kvn_g, kvn_b, KVN, DKV, 64, KR, (int)R);

  // kv = LN(kv_n) @ Wukv^T + b
  gemm_ff_wmma<<<dim3(DUKV / 128, (unsigned)(R / 128)), blk256, 0, stream>>>(
      KVN, Wukv_p, Wukv_b, S, DUKV, DUKV, DKV);
  pack_kv_kernel<<<dim3(T_SZ, B_SZ * H_SZ), blk128, 0, stream>>>(S, KR, KF, VT);

  // causal flash attention
  flash_attn_kernel<<<dim3(T_SZ / 64, B_SZ * H_SZ), blk128, 0, stream>>>(QF, KF, VT, Y16);

  // out = y @ Wo^T + b  (writes d_out, f32)
  gemm_ff_wmma<<<dim3(DM / 128, (unsigned)(R / 128)), blk256, 0, stream>>>(
      Y16, Wo_p, Wo_b, out, DM, DM, DM);
}
